// ResBlock_43052752175220
// MI455X (gfx1250) — compile-verified, matrix operand order checked by reading, not run on hardware
//
#include <hip/hip_runtime.h>
#include <hip/hip_bf16.h>
#include <math.h>

typedef __attribute__((ext_vector_type(16))) _Float16 v16h;
typedef __attribute__((ext_vector_type(8)))  _Float16 v8h;
typedef __attribute__((ext_vector_type(8)))  float    v8f;

#define HWSZ 1024
#define NB   16
#define TMM  128     // macro tile M
#define TNN  64      // macro tile N
#define TK   32

// ---------------- CDNA5 async-to-LDS helpers (gfx1250) ----------------
// Generic pointers to __shared__ carry the LDS byte offset in addr[31:0]
// (ISA 10.2 aperture mapping), so the low dword is a valid per-lane LDS
// destination address for GLOBAL_LOAD_ASYNC_TO_LDS_*.
__device__ __forceinline__ void async_b128(_Float16* lds_dst, const _Float16* gsrc) {
  unsigned l = (unsigned)(uintptr_t)lds_dst;
  asm volatile("global_load_async_to_lds_b128 %0, %1, off"
               :: "v"(l), "v"(gsrc) : "memory");
}

// ---------------- elementwise / reduction kernels ----------------

// 2D convert f32->f16 with zero padding: src [M,K] -> dst [Mp,Kp]
__global__ __launch_bounds__(256) void k_cvt_pad2d(const float* __restrict__ s,
                                                   _Float16* __restrict__ d,
                                                   int M, int K, int Mp, int Kp) {
  int i = blockIdx.x * 256 + threadIdx.x;
  if (i >= Mp * Kp) return;
  int m = i / Kp, k = i % Kp;
  d[i] = (m < M && k < K) ? (_Float16)s[(size_t)m * K + k] : (_Float16)0.f;
}

// [B,Ci,HW] f32 -> [B,Cp,HW] f16, zero channels >= Ci
__global__ __launch_bounds__(256) void k_cvt_pad_chan(const float* __restrict__ s,
                                                      _Float16* __restrict__ d,
                                                      int Ci, int Cp) {
  int i = blockIdx.x * 256 + threadIdx.x;
  if (i >= NB * Cp * HWSZ) return;
  int j = i & (HWSZ - 1);
  int c = (i >> 10) % Cp;
  int b = i / (Cp << 10);
  d[i] = (c < Ci) ? (_Float16)s[((size_t)b * Ci + c) * HWSZ + j] : (_Float16)0.f;
}

__global__ __launch_bounds__(256) void k_add(const float* __restrict__ a,
                                             const float* __restrict__ b,
                                             float* __restrict__ o, int n) {
  int i = blockIdx.x * 256 + threadIdx.x;
  if (i < n) o[i] = a[i] + b[i];
}

// x layout [B, C, HW]; one block per channel
__global__ __launch_bounds__(256) void k_bn_stats(const float* __restrict__ x, int C,
                                                  float* __restrict__ mean,
                                                  float* __restrict__ rstd) {
  int c = blockIdx.x, tid = threadIdx.x;
  float s = 0.f, s2 = 0.f;
  for (int i = tid; i < NB * HWSZ; i += 256) {
    int b = i >> 10, j = i & (HWSZ - 1);
    float v = x[((size_t)b * C + c) * HWSZ + j];
    s += v; s2 += v * v;
  }
  __shared__ float sh0[256], sh1[256];
  sh0[tid] = s; sh1[tid] = s2; __syncthreads();
  for (int st = 128; st > 0; st >>= 1) {
    if (tid < st) { sh0[tid] += sh0[tid + st]; sh1[tid] += sh1[tid + st]; }
    __syncthreads();
  }
  if (tid == 0) {
    float n = (float)(NB * HWSZ);
    float m = sh0[0] / n;
    float var = sh1[0] / n - m * m;
    mean[c] = m;
    rstd[c] = rsqrtf(var + 1e-5f);
  }
}

__global__ __launch_bounds__(256) void k_bn_relu_f16(const float* __restrict__ x,
    const float* __restrict__ mean, const float* __restrict__ rstd,
    const float* __restrict__ g, const float* __restrict__ bb,
    _Float16* __restrict__ y, int C, int total, int relu) {
  int i = blockIdx.x * 256 + threadIdx.x;
  if (i >= total) return;
  int c = (i >> 10) % C;
  float v = (x[i] - mean[c]) * rstd[c] * g[c] + bb[c];
  if (relu) v = fmaxf(v, 0.f);
  y[i] = (_Float16)v;
}

// one image: src [C,32,32] f16 -> colT [1024, Kp] f16 (K-contiguous rows),
// pad=1; K entries >= C*9 zeroed
__global__ __launch_bounds__(256) void k_im2col_t(const _Float16* __restrict__ src,
                                                  _Float16* __restrict__ colt,
                                                  int C, int Kp) {
  int i = blockIdx.x * 256 + threadIdx.x;
  if (i >= Kp * HWSZ) return;
  int r = i % Kp;
  int n = i / Kp;
  _Float16 v = (_Float16)0.f;
  if (r < C * 9) {
    int kx = r % 3, ky = (r / 3) % 3, c = r / 9;
    int xx = n & 31, yy = n >> 5;
    int sy = yy + ky - 1, sx = xx + kx - 1;
    if (sy >= 0 && sy < 32 && sx >= 0 && sx < 32)
      v = src[((size_t)c << 10) + (sy << 5) + sx];
  }
  colt[i] = v;
}

// one block per row of S [1024,1024]; write probs as f16
__global__ __launch_bounds__(256) void k_softmax_f16(const float* __restrict__ S,
                                                     _Float16* __restrict__ P) {
  int row = blockIdx.x, tid = threadIdx.x;
  const float* r = S + (size_t)row * HWSZ;
  __shared__ float sh[256];
  float mx = -3.4e38f;
  for (int i = tid; i < HWSZ; i += 256) mx = fmaxf(mx, r[i]);
  sh[tid] = mx; __syncthreads();
  for (int st = 128; st > 0; st >>= 1) { if (tid < st) sh[tid] = fmaxf(sh[tid], sh[tid + st]); __syncthreads(); }
  mx = sh[0]; __syncthreads();
  float s = 0.f;
  for (int i = tid; i < HWSZ; i += 256) s += __expf(r[i] - mx);
  sh[tid] = s; __syncthreads();
  for (int st = 128; st > 0; st >>= 1) { if (tid < st) sh[tid] += sh[tid + st]; __syncthreads(); }
  float inv = 1.f / sh[0];
  for (int i = tid; i < HWSZ; i += 256)
    P[(size_t)row * HWSZ + i] = (_Float16)(__expf(r[i] - mx) * inv);
}

// ---------------- WMMA GEMM (double-buffered async staging) ----------------
// C[M,N] = alpha * A_op x B_op (+bias[m]) (+resid[m,n])
// A_op[m,k] = TA ? A[k*lda+m] : A[m*lda+k]
// B_op[k,n] = TB ? B[n*ldb+k] : B[k*ldb+n]
// Contiguous staging (!TA for A, TB for B) uses GLOBAL_LOAD_ASYNC_TO_LDS_B128,
// double-buffered so chunk i+1's DMA overlaps chunk i's WMMAs; ASYNCcnt's
// in-order completion lets "s_wait_asynccnt <= just-issued" guarantee the
// current buffer has landed.
// Host-side padding contracts: K%32==0, N%64==0, lda%8==0, ldb%8==0,
// A readable for ceil128(M) rows (rows>=M garbage OK; K-pad rows must be zero
// on the operand that carries real data along K).
template <int TA, int TB, int BIAS, int RES, int OUTH>
__global__ __launch_bounds__(256) void k_gemm_wmma(
    const _Float16* __restrict__ A, const _Float16* __restrict__ B,
    float* __restrict__ Cf, _Float16* __restrict__ Ch,
    const float* __restrict__ bias, const float* __restrict__ resid,
    int M, int N, int K, int lda, int ldb, int ldc, float alpha) {
  __shared__ __align__(16) _Float16 As [2][TMM][TK + 8];  // A row-major  [m][k]
  __shared__ __align__(16) _Float16 BsT[2][TNN][TK + 8];  // B transposed [n][k]
  int tid = threadIdx.x;
  int lane = tid & 31, wave = tid >> 5;
  int wm = wave & 3, wn = wave >> 2;            // 4 waves on M (32 rows each), 2 on N
  int m0 = blockIdx.y * TMM, n0 = blockIdx.x * TNN;
  v8f acc[2][2] = {};                           // 32x32 output per wave
  int mr0   = wm * 32 + (lane & 15);
  int nc0   = wn * 32 + (lane & 15);
  int khalf = (lane >> 4) << 3;                 // lanes 16-31: +8 in K
  // staging map: 256 threads = 64 rows x 4 eight-wide K chunks (A does 2 row groups)
  int sr = tid & 63, skc = (tid >> 6) << 3;
  constexpr int NASY = (TA ? 0 : 2) + (TB ? 1 : 0);  // asyncs per stage call

  auto stage = [&](int buf, int k0) {
    if (!TA) {
#pragma unroll
      for (int gr = 0; gr < 2; ++gr) {
        int r = sr + gr * 64;
        async_b128(&As[buf][r][skc], A + (size_t)(m0 + r) * lda + (k0 + skc));
      }
    } else {
#pragma unroll
      for (int gr = 0; gr < 2; ++gr) {
        int r = sr + gr * 64;
        v8h t;
#pragma unroll
        for (int j = 0; j < 8; ++j)             // lane-coalesced along m
          t[j] = A[(size_t)(k0 + skc + j) * lda + (m0 + r)];
        *(v8h*)&As[buf][r][skc] = t;
      }
      if (k0 + TK < K)
        __builtin_prefetch(A + (size_t)(k0 + TK + skc) * lda + (m0 + sr), 0, 0);
    }
    if (TB) {
      async_b128(&BsT[buf][sr][skc], B + (size_t)(n0 + sr) * ldb + (k0 + skc));
    } else {
      v8h t;
#pragma unroll
      for (int j = 0; j < 8; ++j)               // lane-coalesced along n
        t[j] = B[(size_t)(k0 + skc + j) * ldb + (n0 + sr)];
      *(v8h*)&BsT[buf][sr][skc] = t;
      if (k0 + TK < K)
        __builtin_prefetch(B + (size_t)(k0 + TK + skc) * ldb + (n0 + sr), 0, 0);
    }
  };

  int nchunk = K / TK;
  stage(0, 0);
  for (int ic = 0; ic < nchunk; ++ic) {
    int cur = ic & 1;
    if (ic + 1 < nchunk) {
      // overlap next chunk's DMA with this chunk's compute; safe: the trailing
      // barrier of iteration ic-1 ordered all reads of buffer cur^1.
      stage(cur ^ 1, (ic + 1) * TK);
      if (NASY)
        asm volatile("s_wait_asynccnt %0" :: "i"(NASY) : "memory");
    } else {
      if (NASY)
        asm volatile("s_wait_asynccnt 0x0" ::: "memory");
    }
    __syncthreads();
    // ---- fragments: two ds_load_b128 per operand fragment (ISA 7.12.2) ----
    v8h a0lo = *(const v8h*)&As [cur][mr0     ][khalf];
    v8h a0hi = *(const v8h*)&As [cur][mr0     ][16 + khalf];
    v8h a1lo = *(const v8h*)&As [cur][mr0 + 16][khalf];
    v8h a1hi = *(const v8h*)&As [cur][mr0 + 16][16 + khalf];
    v8h b0lo = *(const v8h*)&BsT[cur][nc0     ][khalf];
    v8h b0hi = *(const v8h*)&BsT[cur][nc0     ][16 + khalf];
    v8h b1lo = *(const v8h*)&BsT[cur][nc0 + 16][khalf];
    v8h b1hi = *(const v8h*)&BsT[cur][nc0 + 16][16 + khalf];
    v16h a0 = __builtin_shufflevector(a0lo, a0hi, 0,1,2,3,4,5,6,7,8,9,10,11,12,13,14,15);
    v16h a1 = __builtin_shufflevector(a1lo, a1hi, 0,1,2,3,4,5,6,7,8,9,10,11,12,13,14,15);
    v16h b0 = __builtin_shufflevector(b0lo, b0hi, 0,1,2,3,4,5,6,7,8,9,10,11,12,13,14,15);
    v16h b1 = __builtin_shufflevector(b1lo, b1hi, 0,1,2,3,4,5,6,7,8,9,10,11,12,13,14,15);
    acc[0][0] = __builtin_amdgcn_wmma_f32_16x16x32_f16(false, a0, false, b0, (short)0, acc[0][0], false, false);
    acc[0][1] = __builtin_amdgcn_wmma_f32_16x16x32_f16(false, a0, false, b1, (short)0, acc[0][1], false, false);
    acc[1][0] = __builtin_amdgcn_wmma_f32_16x16x32_f16(false, a1, false, b0, (short)0, acc[1][0], false, false);
    acc[1][1] = __builtin_amdgcn_wmma_f32_16x16x32_f16(false, a1, false, b1, (short)0, acc[1][1], false, false);
    __syncthreads();
  }

  // C/D layout: VGPR i -> M = i + (lane>=16 ? 8 : 0), N = lane&15
  int mco = (lane >> 4) << 3;
#pragma unroll
  for (int am = 0; am < 2; ++am) {
#pragma unroll
    for (int t = 0; t < 2; ++t) {
      int nn = n0 + wn * 32 + t * 16 + (lane & 15);
      if (nn >= N) continue;
#pragma unroll
      for (int i = 0; i < 8; ++i) {
        int mm = m0 + wm * 32 + am * 16 + mco + i;
        if (mm >= M) continue;
        float v = alpha * acc[am][t][i];
        if (BIAS) v += bias[mm];
        if (RES)  v += resid[(size_t)mm * ldc + nn];
        if (OUTH) Ch[(size_t)mm * ldc + nn] = (_Float16)v;
        else      Cf[(size_t)mm * ldc + nn] = v;
      }
    }
  }
}

// ---------------- host orchestration ----------------

static inline int r32(int x) { return (x + 31) & ~31; }

extern "C" void kernel_launch(void* const* d_in, const int* in_sizes, int n_in,
                              void* d_out, int out_size, void* d_ws, size_t ws_size,
                              hipStream_t stream) {
  (void)in_sizes; (void)n_in; (void)out_size; (void)ws_size;
  const int  CIN[4]  = {3, 128, 128, 256};
  const int  COUT[4] = {128, 128, 256, 256};
  const bool HSC[4]  = {true, false, true, false};

  int gi = 0;
  const float* X = (const float*)d_in[gi++];
  struct Blk {
    const float *bn1_g,*bn1_b,*w1,*b1,*bn2_g,*bn2_b,*w2,*b2;
    const float *abn_g,*abn_b,*wq,*bq,*wk,*bk,*wv,*bv,*wo,*bo,*scw,*scb;
  } bp[4];
  for (int i = 0; i < 4; i++) {
    Blk& p = bp[i];
    p.bn1_g=(const float*)d_in[gi++]; p.bn1_b=(const float*)d_in[gi++];
    p.w1   =(const float*)d_in[gi++]; p.b1   =(const float*)d_in[gi++];
    p.bn2_g=(const float*)d_in[gi++]; p.bn2_b=(const float*)d_in[gi++];
    p.w2   =(const float*)d_in[gi++]; p.b2   =(const float*)d_in[gi++];
    p.abn_g=(const float*)d_in[gi++]; p.abn_b=(const float*)d_in[gi++];
    p.wq=(const float*)d_in[gi++]; p.bq=(const float*)d_in[gi++];
    p.wk=(const float*)d_in[gi++]; p.bk=(const float*)d_in[gi++];
    p.wv=(const float*)d_in[gi++]; p.bv=(const float*)d_in[gi++];
    p.wo=(const float*)d_in[gi++]; p.bo=(const float*)d_in[gi++];
    if (HSC[i]) { p.scw=(const float*)d_in[gi++]; p.scb=(const float*)d_in[gi++]; }
    else        { p.scw=nullptr; p.scb=nullptr; }
  }
  const float* tbn_g=(const float*)d_in[gi++];
  const float* tbn_b=(const float*)d_in[gi++];
  const float* twp  =(const float*)d_in[gi++];
  const float* tbp  =(const float*)d_in[gi++];

  // carve workspace
  uintptr_t wp = (uintptr_t)d_ws;
  auto alloc = [&](size_t bytes) -> void* {
    uintptr_t p = (wp + 255) & ~(uintptr_t)255;
    wp = p + bytes;
    return (void*)p;
  };
  const size_t ACT = (size_t)NB * 256 * HWSZ;
  float*    A0  = (float*)alloc(ACT * 4);
  float*    A1  = (float*)alloc(ACT * 4);
  float*    A2  = (float*)alloc(ACT * 4);
  _Float16* T   = (_Float16*)alloc(ACT * 2);
  _Float16* XH  = (_Float16*)alloc(ACT * 2);
  _Float16* COLT= (_Float16*)alloc((size_t)2304 * HWSZ * 2);   // [1024][Kp]
  _Float16* QH  = (_Float16*)alloc((size_t)256 * HWSZ * 2);
  _Float16* KH  = (_Float16*)alloc((size_t)256 * HWSZ * 2);
  _Float16* VH  = (_Float16*)alloc((size_t)256 * HWSZ * 2);
  _Float16* OH  = (_Float16*)alloc((size_t)256 * HWSZ * 2);
  float*    S   = (float*)alloc((size_t)HWSZ * HWSZ * 4);
  _Float16* P   = (_Float16*)alloc((size_t)HWSZ * HWSZ * 2);
  float*    MEAN= (float*)alloc(256 * 4);
  float*    RSTD= (float*)alloc(256 * 4);

  auto ew_grid = [](size_t n){ return dim3((unsigned)((n + 255) / 256)); };
  auto cvtw = [&](const float* s, _Float16* d, int M, int K, int Mp, int Kp) {
    k_cvt_pad2d<<<ew_grid((size_t)Mp * Kp), dim3(256), 0, stream>>>(s, d, M, K, Mp, Kp);
  };

  // convert all weights to padded f16 once per call (deterministic)
  _Float16 *w1h[4], *w2h[4], *wqh[4], *wkh[4], *wvh[4], *woh[4], *scwh[4], *twh;
  int w1K[4], scK[4];
  for (int i = 0; i < 4; i++) {
    int ci = CIN[i], co = COUT[i];
    w1K[i] = r32(ci * 9);
    scK[i] = r32(ci);
    w1h[i]=(_Float16*)alloc((size_t)co * w1K[i] * 2); cvtw(bp[i].w1, w1h[i], co, ci*9, co, w1K[i]);
    w2h[i]=(_Float16*)alloc((size_t)co * co*9 * 2);   cvtw(bp[i].w2, w2h[i], co, co*9, co, co*9);
    wqh[i]=(_Float16*)alloc((size_t)co * co * 2);     cvtw(bp[i].wq, wqh[i], co, co, co, co);
    wkh[i]=(_Float16*)alloc((size_t)co * co * 2);     cvtw(bp[i].wk, wkh[i], co, co, co, co);
    wvh[i]=(_Float16*)alloc((size_t)co * co * 2);     cvtw(bp[i].wv, wvh[i], co, co, co, co);
    woh[i]=(_Float16*)alloc((size_t)co * co * 2);     cvtw(bp[i].wo, woh[i], co, co, co, co);
    if (bp[i].scw) {
      scwh[i]=(_Float16*)alloc((size_t)co * scK[i] * 2);
      cvtw(bp[i].scw, scwh[i], co, ci, co, scK[i]);
    } else scwh[i]=nullptr;
  }
  // tail weights: pad M 3->128 (macro tile reads 128 A rows), K=2304
  twh=(_Float16*)alloc((size_t)128 * 2304 * 2); cvtw(twp, twh, 3, 2304, 128, 2304);

  auto gdim = [](int M, int N){
    return dim3((unsigned)((N + TNN - 1) / TNN), (unsigned)((M + TMM - 1) / TMM));
  };
  // the five GEMM configurations used by the network:
  auto gemm_conv = [&](const _Float16* A, const _Float16* Bt, float* C, const float* bias,
                       int M, int K) {   // B = COLT [1024][K], out [M][1024]
    k_gemm_wmma<0,1,1,0,0><<<gdim(M,HWSZ), dim3(256), 0, stream>>>(A, Bt, C, nullptr, bias, nullptr, M, HWSZ, K, K, K, HWSZ, 1.f);
  };
  auto gemm_cr = [&](const _Float16* A, const _Float16* B, float* C, const float* bias,
                     const float* resid, int M, int N, int K, int lda, int ldb, int ldc) {
    k_gemm_wmma<0,0,1,1,0><<<gdim(M,N), dim3(256), 0, stream>>>(A, B, C, nullptr, bias, resid, M, N, K, lda, ldb, ldc, 1.f);
  };
  auto gemm_ch = [&](const _Float16* A, const _Float16* B, _Float16* C, const float* bias,
                     int M, int N, int K, int lda, int ldb, int ldc) {
    k_gemm_wmma<0,0,1,0,1><<<gdim(M,N), dim3(256), 0, stream>>>(A, B, nullptr, C, bias, nullptr, M, N, K, lda, ldb, ldc, 1.f);
  };
  auto gemm_s = [&](const _Float16* A, const _Float16* B, float* C,
                    int M, int N, int K, int lda, int ldb, int ldc, float alpha) {
    k_gemm_wmma<1,0,0,0,0><<<gdim(M,N), dim3(256), 0, stream>>>(A, B, C, nullptr, nullptr, nullptr, M, N, K, lda, ldb, ldc, alpha);
  };
  auto gemm_o = [&](const _Float16* A, const _Float16* B, _Float16* C,
                    int M, int N, int K, int lda, int ldb, int ldc) {
    k_gemm_wmma<0,1,0,0,1><<<gdim(M,N), dim3(256), 0, stream>>>(A, B, nullptr, C, nullptr, nullptr, M, N, K, lda, ldb, ldc, 1.f);
  };

  const float* cur = X;
  for (int bi = 0; bi < 4; bi++) {
    const Blk& p = bp[bi];
    int Ci = CIN[bi], Co = COUT[bi];
    size_t nI = (size_t)NB * Ci * HWSZ, nO = (size_t)NB * Co * HWSZ;
    int K1 = w1K[bi];                 // padded C_in*9
    int K2 = Co * 9;

    // BN1 + ReLU -> T (f16)
    k_bn_stats<<<dim3(Ci), dim3(256), 0, stream>>>(cur, Ci, MEAN, RSTD);
    k_bn_relu_f16<<<ew_grid(nI), dim3(256), 0, stream>>>(cur, MEAN, RSTD, p.bn1_g, p.bn1_b, T, Ci, (int)nI, 1);
    // conv1 (3x3) -> A1
    for (int b = 0; b < NB; b++) {
      k_im2col_t<<<ew_grid((size_t)K1*HWSZ), dim3(256), 0, stream>>>(T + (size_t)b*Ci*HWSZ, COLT, Ci, K1);
      gemm_conv(w1h[bi], COLT, A1 + (size_t)b*Co*HWSZ, p.b1, Co, K1);
    }
    // BN2 + ReLU -> T
    k_bn_stats<<<dim3(Co), dim3(256), 0, stream>>>(A1, Co, MEAN, RSTD);
    k_bn_relu_f16<<<ew_grid(nO), dim3(256), 0, stream>>>(A1, MEAN, RSTD, p.bn2_g, p.bn2_b, T, Co, (int)nO, 1);
    // conv2 (3x3) -> A1 (A1 fully consumed above)
    for (int b = 0; b < NB; b++) {
      k_im2col_t<<<ew_grid((size_t)K2*HWSZ), dim3(256), 0, stream>>>(T + (size_t)b*Co*HWSZ, COLT, Co, K2);
      gemm_conv(w2h[bi], COLT, A1 + (size_t)b*Co*HWSZ, p.b2, Co, K2);
    }
    // residual (+1x1 shortcut when Ci != Co) -> A2
    if (p.scw) {
      int Cp = scK[bi];
      k_cvt_pad_chan<<<ew_grid((size_t)NB*Cp*HWSZ), dim3(256), 0, stream>>>(cur, XH, Ci, Cp);
      for (int b = 0; b < NB; b++)
        gemm_cr(scwh[bi], XH + (size_t)b*Cp*HWSZ, A2 + (size_t)b*Co*HWSZ, p.scb,
                A1 + (size_t)b*Co*HWSZ, Co, HWSZ, Cp, Cp, HWSZ, HWSZ);
    } else {
      k_add<<<ew_grid(nO), dim3(256), 0, stream>>>(A1, cur, A2, (int)nO);
    }

    // attention: BN -> T (f16), then per image q/k/v, scores, softmax, out
    k_bn_stats<<<dim3(Co), dim3(256), 0, stream>>>(A2, Co, MEAN, RSTD);
    k_bn_relu_f16<<<ew_grid(nO), dim3(256), 0, stream>>>(A2, MEAN, RSTD, p.abn_g, p.abn_b, T, Co, (int)nO, 0);
    float ascale = 1.f / sqrtf((float)Co);
    for (int b = 0; b < NB; b++) {
      const _Float16* hb = T + (size_t)b*Co*HWSZ;
      gemm_ch(wqh[bi], hb, QH, p.bq, Co, HWSZ, Co, Co, HWSZ, HWSZ);
      gemm_ch(wkh[bi], hb, KH, p.bk, Co, HWSZ, Co, Co, HWSZ, HWSZ);
      gemm_ch(wvh[bi], hb, VH, p.bv, Co, HWSZ, Co, Co, HWSZ, HWSZ);
      // S[q,k] = scale * sum_c Q[c,q] K[c,k]   (A transposed)
      gemm_s(QH, KH, S, HWSZ, HWSZ, Co, HWSZ, HWSZ, HWSZ, ascale);
      k_softmax_f16<<<dim3(HWSZ), dim3(256), 0, stream>>>(S, P);
      // O[c,q] = sum_k V[c,k] P[q,k]           (B transposed)
      gemm_o(VH, P, OH, Co, HWSZ, HWSZ, HWSZ, HWSZ, HWSZ);
      // x_out = wo*O + bo + x_new  -> A0
      gemm_cr(woh[bi], OH, A0 + (size_t)b*Co*HWSZ, p.bo, A2 + (size_t)b*Co*HWSZ,
              Co, HWSZ, Co, Co, HWSZ, HWSZ);
    }
    cur = A0;
  }

  // tail: BN+ReLU -> conv3x3 to 3 channels -> d_out
  k_bn_stats<<<dim3(256), dim3(256), 0, stream>>>(cur, 256, MEAN, RSTD);
  k_bn_relu_f16<<<ew_grid((size_t)NB*256*HWSZ), dim3(256), 0, stream>>>(cur, MEAN, RSTD, tbn_g, tbn_b, T, 256, NB*256*HWSZ, 1);
  float* OUT = (float*)d_out;
  for (int b = 0; b < NB; b++) {
    k_im2col_t<<<ew_grid((size_t)2304*HWSZ), dim3(256), 0, stream>>>(T + (size_t)b*256*HWSZ, COLT, 256, 2304);
    gemm_conv(twh, COLT, OUT + (size_t)b*3*HWSZ, tbp, 3, 2304);
  }
}
